// EncoderModel_16166256902694
// MI455X (gfx1250) — matrix operand order repaired and checked by reading, not compiled, not run on hardware
//
#include <hip/hip_runtime.h>
#include <hip/hip_bf16.h>

// ---------------- model constants ----------------
constexpr int V_  = 32000;
constexpr int D_  = 1024;
constexpr int NL_ = 6;
constexpr int H_  = 16;
constexpr int DK_ = 64;
constexpr int DF_ = 4096;
constexpr int S_  = 1024;
constexpr int B_  = 4;
constexpr int M_  = B_ * S_;        // 4096 rows of activations

typedef __attribute__((ext_vector_type(16))) _Float16 v16h;
typedef __attribute__((ext_vector_type(4)))  _Float16 h4;
typedef __attribute__((ext_vector_type(2)))  _Float16 h2;
typedef __attribute__((ext_vector_type(8)))  float    v8f;

// ---------------- WMMA GEMM: C = act(A(MxK) * B(KxN) + bias) ----------------
// 128x128 tile per 256-thread block (8 wave32s in a 4x2 grid, 32x64 per wave).
// Double-buffered LDS + register-staged global loads: two barriers per K-step,
// HBM latency hidden behind the 8 WMMAs of the previous step.
#define TILE_M 128
#define TILE_N 128
#define TILE_K 32

template <int ACT>
__global__ __launch_bounds__(256) void gemm_bias_act_kernel(
    const float* __restrict__ A, int lda,
    const float* __restrict__ Bm, int ldb,
    const float* __restrict__ bias,
    float* __restrict__ C, int ldc,
    int K)
{
  __shared__ _Float16 As[2][TILE_M * TILE_K];   // [buf][row][k]
  __shared__ _Float16 Bs[2][TILE_N * TILE_K];   // [buf][col][k]  (K-transposed)

  const int tid  = threadIdx.x;
  const int wave = tid >> 5;
  const int lane = tid & 31;
  const int wm   = wave >> 1;                // 0..3 -> 32-row slab
  const int wn   = wave & 1;                 // 0..1 -> 64-col slab

  const int m0 = blockIdx.y * TILE_M;
  const int n0 = blockIdx.x * TILE_N;

  v8f acc[2][4] = {};

  const int ar = tid >> 1;                   // A stage: row 0..127
  const int ac = (tid & 1) * 16;             // A stage: k base
  const int kp = tid >> 4;                   // B stage: k-pair 0..15 -> k = 2kp, 2kp+1
  const int bn = (tid & 15) * 8;             // B stage: n base (8 cols)

  const float* aptr  = A + (size_t)(m0 + ar) * lda + ac;       // + k0 per step
  const float* bptr0 = Bm + (size_t)(2 * kp) * ldb + n0 + bn;  // + k0*ldb per step
  const float* bptr1 = bptr0 + ldb;

  // ---- prologue: preload K-tile 0 into registers ----
  float4 ra[4], rb0[2], rb1[2];
  #pragma unroll
  for (int j = 0; j < 4; ++j) ra[j] = ((const float4*)aptr)[j];
  #pragma unroll
  for (int j = 0; j < 2; ++j) rb0[j] = ((const float4*)bptr0)[j];
  #pragma unroll
  for (int j = 0; j < 2; ++j) rb1[j] = ((const float4*)bptr1)[j];

  int buf = 0;
  for (int k0 = 0; k0 < K; k0 += TILE_K) {
    // ---- store staged registers (f32 -> f16) into LDS[buf] ----
    #pragma unroll
    for (int j = 0; j < 4; ++j) {
      h4 h;
      h[0] = (_Float16)ra[j].x; h[1] = (_Float16)ra[j].y;
      h[2] = (_Float16)ra[j].z; h[3] = (_Float16)ra[j].w;
      *(h4*)&As[buf][ar * TILE_K + ac + j * 4] = h;       // ds_store_b64
    }
    {
      const float* f0 = (const float*)rb0;   // 8 floats, K-row 2kp
      const float* f1 = (const float*)rb1;   // 8 floats, K-row 2kp+1
      #pragma unroll
      for (int j = 0; j < 8; ++j) {          // pack (k,k+1) -> ds_store_b32
        h2 h;
        h[0] = (_Float16)f0[j];
        h[1] = (_Float16)f1[j];
        *(h2*)&Bs[buf][(bn + j) * TILE_K + 2 * kp] = h;
      }
    }
    __syncthreads();

    // ---- issue next K-tile's global loads; they complete under the WMMAs ----
    if (k0 + TILE_K < K) {
      const float* an = aptr + (k0 + TILE_K);
      const float* b0 = bptr0 + (size_t)(k0 + TILE_K) * ldb;
      const float* b1 = bptr1 + (size_t)(k0 + TILE_K) * ldb;
      #pragma unroll
      for (int j = 0; j < 4; ++j) ra[j] = ((const float4*)an)[j];
      #pragma unroll
      for (int j = 0; j < 2; ++j) rb0[j] = ((const float4*)b0)[j];
      #pragma unroll
      for (int j = 0; j < 2; ++j) rb1[j] = ((const float4*)b1)[j];
      if (k0 + 2 * TILE_K < K) {               // pull tile k0+64 toward GL2
        __builtin_prefetch(aptr + (k0 + 2 * TILE_K), 0, 0);
        __builtin_prefetch(bptr0 + (size_t)(k0 + 2 * TILE_K) * ldb, 0, 0);
      }
    }

    // ---- A fragments (ISA 16-bit A 16x32 layout) ----
    v16h afrag[2];
    #pragma unroll
    for (int mi = 0; mi < 2; ++mi) {
      int arow = wm * 32 + mi * 16 + (lane & 15);
      int kb = (lane < 16) ? 0 : 8;
      const _Float16* s = &As[buf][arow * TILE_K];
      #pragma unroll
      for (int j = 0; j < 8; ++j) {
        afrag[mi][j]     = s[kb + j];          // K = kb .. kb+7
        afrag[mi][8 + j] = s[kb + 16 + j];     // K = kb+16 .. kb+23
      }
    }
    // ---- B fragments: lane = column, K = kb..kb+7, kb+16..kb+23 ----
    v16h bfrag[4];
    #pragma unroll
    for (int ni = 0; ni < 4; ++ni) {
      int bcol = wn * 64 + ni * 16 + (lane & 15);
      int kb = (lane < 16) ? 0 : 8;
      const _Float16* s = &Bs[buf][bcol * TILE_K + kb];
      #pragma unroll
      for (int j = 0; j < 8; ++j) {
        bfrag[ni][j]     = s[j];
        bfrag[ni][8 + j] = s[16 + j];
      }
    }

    #pragma unroll
    for (int mi = 0; mi < 2; ++mi)
      #pragma unroll
      for (int ni = 0; ni < 4; ++ni)
        acc[mi][ni] = __builtin_amdgcn_wmma_f32_16x16x32_f16(
            false, afrag[mi], false, bfrag[ni],
            (short)0, acc[mi][ni], false, false);

    buf ^= 1;
    __syncthreads();
  }

  // ---- epilogue: bias + activation + store (C layout per ISA) ----
  #pragma unroll
  for (int mi = 0; mi < 2; ++mi) {
    #pragma unroll
    for (int ni = 0; ni < 4; ++ni) {
      int col = n0 + wn * 64 + ni * 16 + (lane & 15);
      float bv = bias[col];
      #pragma unroll
      for (int r = 0; r < 8; ++r) {
        int row = m0 + wm * 32 + mi * 16 + ((lane < 16) ? r : (8 + r));
        float val = acc[mi][ni][r] + bv;
        if (ACT == 1) val = fmaxf(val, 0.0f);
        C[(size_t)row * ldc + col] = val;
      }
    }
  }
}

// ---------------- fused attention: one wave per (b,h,q) row ----------------
__global__ __launch_bounds__(256) void attn_kernel(
    const float* __restrict__ q, const float* __restrict__ k,
    const float* __restrict__ v, const int* __restrict__ mask,
    float* __restrict__ ctx)
{
  __shared__ float qs[8][DK_];
  __shared__ float sc[8][S_];

  const int w    = threadIdx.x >> 5;
  const int lane = threadIdx.x & 31;
  const int gr   = blockIdx.x * 8 + w;          // 0 .. B*H*S-1
  const int b    = gr / (H_ * S_);
  const int rem  = gr % (H_ * S_);
  const int h    = rem / S_;
  const int qi   = rem % S_;

  const size_t rowbase = (size_t)(b * S_ + qi) * D_ + h * DK_;
  qs[w][lane * 2]     = q[rowbase + lane * 2];
  qs[w][lane * 2 + 1] = q[rowbase + lane * 2 + 1];
  __syncthreads();

  // pass 1: scores + local max
  float m = -1e30f;
  for (int c = 0; c < 32; ++c) {
    int kk = c * 32 + lane;
    const float* kr = k + (size_t)(b * S_ + kk) * D_ + h * DK_;
    float s = 0.0f;
    #pragma unroll
    for (int d = 0; d < DK_; ++d) s += qs[w][d] * kr[d];
    s *= 0.125f;                                 // 1/sqrt(64)
    if (mask[b * S_ + kk] == 0) s = -1e9f;
    sc[w][kk] = s;
    m = fmaxf(m, s);
  }
  #pragma unroll
  for (int off = 16; off > 0; off >>= 1) m = fmaxf(m, __shfl_xor(m, off, 32));
  __syncthreads();

  // pass 2: exp + sum
  float sum = 0.0f;
  for (int c = 0; c < 32; ++c) {
    int kk = c * 32 + lane;
    float p = __expf(sc[w][kk] - m);
    sc[w][kk] = p;
    sum += p;
  }
  #pragma unroll
  for (int off = 16; off > 0; off >>= 1) sum += __shfl_xor(sum, off, 32);
  float inv = 1.0f / sum;
  __syncthreads();

  // pass 3: out[d] = sum_k p_k * V[k][d], lanes split d
  const int d0 = lane * 2;
  float a0 = 0.0f, a1 = 0.0f;
  for (int kk = 0; kk < S_; ++kk) {
    float p = sc[w][kk];
    const float* vr = v + (size_t)(b * S_ + kk) * D_ + h * DK_;
    a0 += p * vr[d0];
    a1 += p * vr[d0 + 1];
  }
  ctx[rowbase + d0]     = a0 * inv;
  ctx[rowbase + d0 + 1] = a1 * inv;
}

// ---------------- residual add + LayerNorm (one block per row) ----------------
__global__ __launch_bounds__(256) void add_ln_kernel(
    const float* __restrict__ x, const float* __restrict__ r,
    const float* __restrict__ g, const float* __restrict__ bb,
    float* __restrict__ out)
{
  __shared__ float sbuf[8];
  const int row = blockIdx.x;
  const int t   = threadIdx.x;
  const float* xr = x + (size_t)row * D_;
  const float* rr = r ? r + (size_t)row * D_ : nullptr;

  float vals[4];
  float s = 0.0f;
  #pragma unroll
  for (int j = 0; j < 4; ++j) {
    int d = t * 4 + j;
    float val = xr[d] + (rr ? rr[d] : 0.0f);
    vals[j] = val;
    s += val;
  }
  #pragma unroll
  for (int off = 16; off > 0; off >>= 1) s += __shfl_xor(s, off, 32);
  if ((t & 31) == 0) sbuf[t >> 5] = s;
  __syncthreads();
  float total = 0.0f;
  #pragma unroll
  for (int wv = 0; wv < 8; ++wv) total += sbuf[wv];
  float mean = total * (1.0f / (float)D_);
  __syncthreads();

  float sq = 0.0f;
  #pragma unroll
  for (int j = 0; j < 4; ++j) { float dv = vals[j] - mean; sq += dv * dv; }
  #pragma unroll
  for (int off = 16; off > 0; off >>= 1) sq += __shfl_xor(sq, off, 32);
  if ((t & 31) == 0) sbuf[t >> 5] = sq;
  __syncthreads();
  float tot2 = 0.0f;
  #pragma unroll
  for (int wv = 0; wv < 8; ++wv) tot2 += sbuf[wv];
  float invsd = rsqrtf(tot2 * (1.0f / (float)D_) + 1e-5f);

  #pragma unroll
  for (int j = 0; j < 4; ++j) {
    int d = t * 4 + j;
    out[(size_t)row * D_ + d] = (vals[j] - mean) * invsd * g[d] + bb[d];
  }
}

// ---------------- embedding * sqrt(D) + positional encoding ----------------
__global__ __launch_bounds__(256) void embed_kernel(
    const int* __restrict__ src, const float* __restrict__ emb,
    float* __restrict__ x)
{
  const int row  = blockIdx.x;          // 0 .. B*S-1
  const int spos = row % S_;
  const int tok  = src[row];
  const int t    = threadIdx.x;
  #pragma unroll
  for (int j = 0; j < 4; ++j) {
    int d = t * 4 + j;
    int half = d >> 1;
    float freq = __expf(-(float)(2 * half) * (9.210340371976184f / (float)D_));
    float ang  = (float)spos * freq;
    float pe   = (d & 1) ? __cosf(ang) : __sinf(ang);
    x[(size_t)row * D_ + d] = emb[(size_t)tok * D_ + d] * 32.0f + pe;
  }
}

// ---------------- host-side orchestration ----------------
static inline void launch_gemm(const float* A, const float* Bm, const float* bias,
                               float* C, int M, int N, int K, int act,
                               hipStream_t stream)
{
  dim3 grid(N / TILE_N, M / TILE_M);
  if (act)
    gemm_bias_act_kernel<1><<<grid, 256, 0, stream>>>(A, K, Bm, N, bias, C, N, K);
  else
    gemm_bias_act_kernel<0><<<grid, 256, 0, stream>>>(A, K, Bm, N, bias, C, N, K);
}

extern "C" void kernel_launch(void* const* d_in, const int* in_sizes, int n_in,
                              void* d_out, int out_size, void* d_ws, size_t ws_size,
                              hipStream_t stream) {
  (void)in_sizes; (void)n_in; (void)out_size; (void)ws_size;

  const int*   src   = (const int*)d_in[0];
  const int*   mask  = (const int*)d_in[1];
  const float* emb   = (const float*)d_in[2];
  const float* Wq    = (const float*)d_in[3];
  const float* bq    = (const float*)d_in[4];
  const float* Wk    = (const float*)d_in[5];
  const float* bk    = (const float*)d_in[6];
  const float* Wv    = (const float*)d_in[7];
  const float* bv    = (const float*)d_in[8];
  const float* Wo    = (const float*)d_in[9];
  const float* bo    = (const float*)d_in[10];
  const float* ln1_g = (const float*)d_in[11];
  const float* ln1_b = (const float*)d_in[12];
  const float* W1    = (const float*)d_in[13];
  const float* b1    = (const float*)d_in[14];
  const float* W2    = (const float*)d_in[15];
  const float* b2    = (const float*)d_in[16];
  const float* ln2_g = (const float*)d_in[17];
  const float* ln2_b = (const float*)d_in[18];
  const float* lnf_g = (const float*)d_in[19];
  const float* lnf_b = (const float*)d_in[20];
  const float* Wgen  = (const float*)d_in[21];
  const float* bgen  = (const float*)d_in[22];
  float* out = (float*)d_out;

  float* ws = (float*)d_ws;
  const size_t X = (size_t)M_ * D_;            // 4096*1024 floats
  float* xA  = ws;
  float* xB  = ws + X;
  float* qb  = ws + 2 * X;
  float* kb  = ws + 3 * X;
  float* vb  = ws + 4 * X;
  float* cb  = ws + 5 * X;
  float* ffb = ws + 6 * X;                     // 4096*4096 floats

  embed_kernel<<<M_, 256, 0, stream>>>(src, emb, xA);

  for (int i = 0; i < NL_; ++i) {
    const size_t wOff = (size_t)i * D_ * D_;
    const size_t fOff = (size_t)i * D_ * DF_;

    launch_gemm(xA, Wq + wOff, bq + (size_t)i * D_, qb, M_, D_, D_, 0, stream);
    launch_gemm(xA, Wk + wOff, bk + (size_t)i * D_, kb, M_, D_, D_, 0, stream);
    launch_gemm(xA, Wv + wOff, bv + (size_t)i * D_, vb, M_, D_, D_, 0, stream);

    attn_kernel<<<(B_ * H_ * S_) / 8, 256, 0, stream>>>(qb, kb, vb, mask, cb);

    launch_gemm(cb, Wo + wOff, bo + (size_t)i * D_, qb, M_, D_, D_, 0, stream);
    add_ln_kernel<<<M_, 256, 0, stream>>>(xA, qb, ln1_g + (size_t)i * D_,
                                          ln1_b + (size_t)i * D_, xB);

    launch_gemm(xB, W1 + fOff, b1 + (size_t)i * DF_, ffb, M_, DF_, D_, 1, stream);
    launch_gemm(ffb, W2 + fOff, b2 + (size_t)i * D_, qb, M_, D_, DF_, 0, stream);
    add_ln_kernel<<<M_, 256, 0, stream>>>(xB, qb, ln2_g + (size_t)i * D_,
                                          ln2_b + (size_t)i * D_, xA);
  }

  add_ln_kernel<<<M_, 256, 0, stream>>>(xA, nullptr, lnf_g, lnf_b, xB);
  launch_gemm(xB, Wgen, bgen, out, M_, V_, D_, 0, stream);
}
